// SingleHeadAttention_44873818308804
// MI455X (gfx1250) — matrix operand order repaired and checked by reading, not compiled, not run on hardware
//
#include <hip/hip_runtime.h>
#include <hip/hip_bf16.h>

#define BATCH 16
#define SEQ   2048
#define EMB   768
#define HD    64

typedef __attribute__((ext_vector_type(16))) __bf16 bf16x16;
typedef __attribute__((ext_vector_type(8)))  __bf16 bf16x8;
typedef __attribute__((ext_vector_type(8)))  float  f32x8;

__device__ __forceinline__ bf16x16 join16(bf16x8 lo, bf16x8 hi) {
  return __builtin_shufflevector(lo, hi, 0,1,2,3,4,5,6,7,8,9,10,11,12,13,14,15);
}

// B-operand: 16 contiguous bf16 at p (two 16B chunks)
__device__ __forceinline__ bf16x16 load_b_contig(const __bf16* p) {
  bf16x8 lo = *(const bf16x8*)p;
  bf16x8 hi = *(const bf16x8*)(p + 8);
  return join16(lo, hi);
}

// A-operand per-lane halves: elems 0..7 at p, elems 8..15 at p+16
__device__ __forceinline__ bf16x16 load_a_split(const __bf16* p) {
  bf16x8 lo = *(const bf16x8*)p;
  bf16x8 hi = *(const bf16x8*)(p + 16);
  return join16(lo, hi);
}

#define WMMA_BF16(A, B, C) \
  __builtin_amdgcn_wmma_f32_16x16x32_bf16(false, (A), false, (B), (short)0, (C), false, false)

// ---------------------------------------------------------------------------
// Kernel 1: projection  out = x @ W + b   (x fp32 [B*S,768], W fp32 [768,64])
// TRANSPOSED==0 : out bf16 [B*S][64]   (Q, K)
// TRANSPOSED==1 : out bf16 Vt[b][d][s] (V, pre-transposed for P·V B-operand)
// ---------------------------------------------------------------------------
template <int TRANSPOSED>
__global__ __launch_bounds__(256) void qkv_proj_kernel(
    const float* __restrict__ x, const float* __restrict__ W,
    const float* __restrict__ bias, __bf16* __restrict__ out)
{
  __shared__ __align__(128) __bf16 WT[64][48];   // [d][k], k<32 used; padded rows
  const int tid  = threadIdx.x;
  const int wave = tid >> 5, lane = tid & 31;
  const int nloc = lane & 15, hi = lane >> 4;
  const int row0 = blockIdx.x * 128 + wave * 16;   // flattened b*SEQ+s row base

  f32x8 acc[4] = {};

  for (int kk = 0; kk < EMB; kk += 32) {
    __syncthreads();
    // cooperative coalesced load of W^T chunk (fp32 -> bf16)
    #pragma unroll
    for (int i = 0; i < 8; ++i) {
      int e = tid + 256 * i;        // 0..2047
      int k = e >> 6, d = e & 63;
      WT[d][k] = (__bf16)W[(kk + k) * HD + d];
    }
    __syncthreads();

    // A tile: row = row0+nloc, K elems {kb..kb+7, kb+16..kb+23}, kb = hi*8
    const float* xp = x + (size_t)(row0 + nloc) * EMB + kk + hi * 8;
    bf16x16 a;
    #pragma unroll
    for (int j = 0; j < 8; ++j) { a[j] = (__bf16)xp[j]; a[j + 8] = (__bf16)xp[j + 16]; }

    #pragma unroll
    for (int t = 0; t < 4; ++t) {
      bf16x16 bm = load_a_split(&WT[t * 16 + nloc][hi * 8]);
      acc[t] = WMMA_BF16(a, bm, acc[t]);
    }
  }

  // bias + store (branch-free: TRANSPOSED is a compile-time constant)
  const float bb[4] = {bias[nloc], bias[16 + nloc], bias[32 + nloc], bias[48 + nloc]};
  #pragma unroll
  for (int t = 0; t < 4; ++t) {
    #pragma unroll
    for (int r = 0; r < 8; ++r) {
      float v = acc[t][r] + bb[t];
      int row = row0 + r + 8 * hi;       // flattened b*SEQ + s
      int d   = t * 16 + nloc;
      if constexpr (TRANSPOSED == 0) {
        out[(size_t)row * HD + d] = (__bf16)v;
      } else {
        int b = row >> 11, s = row & (SEQ - 1);
        out[((size_t)(b * HD + d) << 11) + s] = (__bf16)v;
      }
    }
  }
}

// ---------------------------------------------------------------------------
// Kernel 2: causal flash attention. One wave = one 32-query tile (two 16-query
// column blocks of the transposed-score WMMAs share every K and V fetch ->
// 32 FLOP per L2 byte). Scores are computed TRANSPOSED (scT = K·Q^T) so the
// f32 accumulator layout coincides element-for-element with the A-operand
// layout required by P·V: probabilities are packed into the next WMMA's A
// operand purely in registers -- no LDS round trip, no fences.
// Qb/Kb bf16 [B*S][64]; Vt bf16 [B][64][S]; out fp32 [B*S][64].
// ---------------------------------------------------------------------------
__global__ __launch_bounds__(256) void flash_attn_kernel(
    const __bf16* __restrict__ Qb, const __bf16* __restrict__ Kb,
    const __bf16* __restrict__ Vt, float* __restrict__ out)
{
  const int tid  = threadIdx.x;
  const int wave = tid >> 5, lane = tid & 31;
  const int nloc = lane & 15, hi = lane >> 4;
  const int tile = blockIdx.x * 8 + wave;          // 0..1023
  const int b    = tile >> 6;
  const int q0   = (tile & 63) * 32;
  const size_t rowbase = (size_t)b * SEQ;

  // Q as B-operands: two query blocks (cols q0+nloc and q0+16+nloc), invariant
  const __bf16* qp0 = Qb + (rowbase + q0 + nloc) * HD + hi * 8;
  const __bf16* qp1 = qp0 + 16 * HD;
  const bf16x16 qb[4] = { load_a_split(qp0), load_a_split(qp0 + 32),     // blk0 d0..31,32..63
                          load_a_split(qp1), load_a_split(qp1 + 32) };   // blk1

  float m[2] = {-3.0e38f, -3.0e38f};   // running max per query block
  float s[2] = {0.0f, 0.0f};           // running denominator per query block
  f32x8 o[8] = {};                     // o[4*blk+t]: (query r+8*hi, dim 16t+nloc)

  for (int jj = 0; jj <= q0 + 16; jj += 32) {      // 32 keys / iteration
    // ---- transposed scores: A = K tiles (M=key), B = Q (N=query) ----
    const __bf16* kp0 = Kb + (rowbase + jj + nloc) * HD + hi * 8;
    const __bf16* kp1 = kp0 + 16 * HD;
    __builtin_prefetch(kp0 + 32 * HD, 0, 3);       // next iteration K (speculative)
    __builtin_prefetch(kp1 + 32 * HD, 0, 3);
    bf16x16 ka0 = load_a_split(kp0);               // keys jj..jj+15,  d 0..31
    bf16x16 ka1 = load_a_split(kp0 + 32);          // keys jj..jj+15,  d 32..63
    bf16x16 ka2 = load_a_split(kp1);               // keys jj+16..+31, d 0..31
    bf16x16 ka3 = load_a_split(kp1 + 32);          // keys jj+16..+31, d 32..63

    f32x8 sc[2][2];                                // [key half][query block]
    sc[0][0] = WMMA_BF16(ka0, qb[0], (f32x8){});
    sc[0][0] = WMMA_BF16(ka1, qb[1], sc[0][0]);
    sc[1][0] = WMMA_BF16(ka2, qb[0], (f32x8){});
    sc[1][0] = WMMA_BF16(ka3, qb[1], sc[1][0]);
    sc[0][1] = WMMA_BF16(ka0, qb[2], (f32x8){});
    sc[0][1] = WMMA_BF16(ka1, qb[3], sc[0][1]);
    sc[1][1] = WMMA_BF16(ka2, qb[2], (f32x8){});
    sc[1][1] = WMMA_BF16(ka3, qb[3], sc[1][1]);

    // ---- per query block: scale, causal mask, online softmax, pack P ----
    bf16x16 pa[2];
    #pragma unroll
    for (int qblk = 0; qblk < 2; ++qblk) {
      const int q = q0 + 16 * qblk + nloc;         // this lane's query
      float mx = -3.0e38f;
      #pragma unroll
      for (int r = 0; r < 8; ++r) {
        const int key = jj + r + 8 * hi;           // sc[0][qblk][r] = (key, q)
        float v0 = sc[0][qblk][r] * 0.125f;
        float v1 = sc[1][qblk][r] * 0.125f;
        v0 = (key      > q) ? -3.0e38f : v0;
        v1 = (key + 16 > q) ? -3.0e38f : v1;
        sc[0][qblk][r] = v0; sc[1][qblk][r] = v1;
        mx = fmaxf(mx, fmaxf(v0, v1));
      }
      mx = fmaxf(mx, __shfl_xor(mx, 16, 32));      // lanes l, l^16 share query
      const float mn    = fmaxf(m[qblk], mx);
      const float alpha = __expf(m[qblk] - mn);
      m[qblk] = mn;

      float ps = 0.0f;
      #pragma unroll
      for (int r = 0; r < 8; ++r) {
        float p0 = __expf(sc[0][qblk][r] - mn);    // key jj + r + 8*hi
        float p1 = __expf(sc[1][qblk][r] - mn);    // key jj + 16 + r + 8*hi
        ps += p0 + p1;
        pa[qblk][r]     = (__bf16)p0;              // A elems 0..7 : keys hi*8+r
        pa[qblk][r + 8] = (__bf16)p1;              // A elems 8..15: keys 16+hi*8+r
      }
      s[qblk] = s[qblk] * alpha + (ps + __shfl_xor(ps, 16, 32));

      // rescale out accumulators (row r+8*hi needs that query's alpha)
      #pragma unroll
      for (int r = 0; r < 8; ++r) {
        float ar = __shfl(alpha, r + 8 * hi, 32);
        o[4 * qblk + 0][r] *= ar; o[4 * qblk + 1][r] *= ar;
        o[4 * qblk + 2][r] *= ar; o[4 * qblk + 3][r] *= ar;
      }
    }

    // ---- accumulate P·V; each V fetch feeds both query blocks ----
    #pragma unroll
    for (int t = 0; t < 4; ++t) {
      const __bf16* vp = Vt + ((size_t)(b * HD + t * 16 + nloc) << 11) + jj + hi * 8;
      __builtin_prefetch(vp + 32, 0, 3);           // next iteration V (speculative)
      bf16x16 vb = load_a_split(vp);
      o[t]     = WMMA_BF16(pa[0], vb, o[t]);
      o[4 + t] = WMMA_BF16(pa[1], vb, o[4 + t]);
    }
  }

  // ---- normalize and store (fp32, 16 lanes x 4B contiguous per row) ----
  #pragma unroll
  for (int qblk = 0; qblk < 2; ++qblk) {
    #pragma unroll
    for (int r = 0; r < 8; ++r) {
      const float inv = 1.0f / __shfl(s[qblk], r + 8 * hi, 32);
      const size_t base = (rowbase + q0 + 16 * qblk + r + 8 * hi) * HD + nloc;
      #pragma unroll
      for (int t = 0; t < 4; ++t)
        out[base + t * 16] = o[4 * qblk + t][r] * inv;
    }
  }
}

// ---------------------------------------------------------------------------
extern "C" void kernel_launch(void* const* d_in, const int* in_sizes, int n_in,
                              void* d_out, int out_size, void* d_ws, size_t ws_size,
                              hipStream_t stream) {
  const float* x  = (const float*)d_in[0];
  const float* Wq = (const float*)d_in[1];
  const float* bq = (const float*)d_in[2];
  const float* Wk = (const float*)d_in[3];
  const float* bk = (const float*)d_in[4];
  const float* Wv = (const float*)d_in[5];
  const float* bv = (const float*)d_in[6];

  const size_t nQKV = (size_t)BATCH * SEQ * HD;    // 2M elems, bf16
  __bf16* Qb = (__bf16*)d_ws;
  __bf16* Kb = Qb + nQKV;
  __bf16* Vt = Kb + nQKV;                          // 12 MB total workspace

  dim3 blk(256);
  dim3 gproj(BATCH * SEQ / 128);                   // 256 blocks
  qkv_proj_kernel<0><<<gproj, blk, 0, stream>>>(x, Wq, bq, Qb);
  qkv_proj_kernel<0><<<gproj, blk, 0, stream>>>(x, Wk, bk, Kb);
  qkv_proj_kernel<1><<<gproj, blk, 0, stream>>>(x, Wv, bv, Vt);

  dim3 gattn(BATCH * SEQ / 32 / 8);                // 128 blocks, 8 waves each
  flash_attn_kernel<<<gattn, blk, 0, stream>>>(Qb, Kb, Vt, (float*)d_out);
}
